// MVGE_45148696215965
// MI455X (gfx1250) — matrix operand order repaired and checked by reading, not compiled, not run on hardware
//
#include <hip/hip_runtime.h>
#include <hip/hip_bf16.h>
#include <stdint.h>

// ---------------------------------------------------------------------------
// MI455X (gfx1250) implementation.
// All GEMMs use v_wmma_f32_16x16x32_bf16 (wave32). z (8 MB) and the scatter
// accumulators (16 MB) are L2-resident (192 MB L2), so fragments are loaded
// straight from global; the only real HBM traffic is the 256 MB output.
// ---------------------------------------------------------------------------

typedef __bf16 v16bf __attribute__((ext_vector_type(16)));
typedef __bf16 v8bf  __attribute__((ext_vector_type(8)));
typedef float  v8f   __attribute__((ext_vector_type(8)));

#define NN 8192      // nodes
#define EE 262144    // edges

// ---- WMMA fragment loaders (contiguous k = 16*(lane/16) + i in both A and B, wave32) ----------

// A: 16x32 bf16 tile from row-major [*, lda]; lane holds row M=lane%16,
// v[i] = K kb+(half?16:0)+i, i = 0..15.
__device__ __forceinline__ v16bf load_a_frag(const __bf16* A, int lda,
                                             int row0, int kb, int lane) {
  int m  = row0 + (lane & 15);
  int k0 = kb + ((lane & 16) ? 16 : 0);
  const __bf16* p = A + (size_t)m * lda + k0;
  v8bf c0 = *(const v8bf*)p;          // 16B
  v8bf c1 = *(const v8bf*)(p + 8);    // 16B
  return __builtin_shufflevector(c0, c1, 0,1,2,3,4,5,6,7,8,9,10,11,12,13,14,15);
}

// B: 32x16 bf16 tile read from N-major storage Bt[N][K]; lane holds col
// N=lane%16, v[i] = K kb+(half?16:0)+i  -> one contiguous 32B load.
__device__ __forceinline__ v16bf load_b_frag(const __bf16* Bt, int ldb,
                                             int col0, int kb, int lane) {
  int n  = col0 + (lane & 15);
  int k0 = kb + ((lane & 16) ? 16 : 0);
  return *(const v16bf*)(Bt + (size_t)n * ldb + k0);
}

__device__ __forceinline__ v8f wmma_bf16(v16bf a, v16bf b, v8f c) {
  return __builtin_amdgcn_wmma_f32_16x16x32_bf16(false, a, false, b,
                                                 (short)0, c, false, false);
}

// ---- Generic bf16 GEMM: C = A0@Bt0^T (+ A1@Bt1^T) + bias, opt relu --------
// Block: 256 thr = 8 waves (4 M x 2 N), wave tile 32x32 -> block tile 128x64.
// A row-major bf16 (lda = K), Bt is N-major bf16 (pre-transposed weights).
__global__ void __launch_bounds__(256)
gemm_bf16_kernel(const __bf16* __restrict__ A0, const __bf16* __restrict__ Bt0,
                 int K0, int ldb0,
                 const __bf16* __restrict__ A1, const __bf16* __restrict__ Bt1,
                 int K1, int ldb1,
                 const float* __restrict__ bias, int act,
                 float* __restrict__ outF, __bf16* __restrict__ outB, int ldOut)
{
  const int lane = threadIdx.x & 31;
  const int wave = threadIdx.x >> 5;
  const int row0 = blockIdx.x * 128 + (wave & 3) * 32;
  const int col0 = blockIdx.y * 64  + (wave >> 2) * 32;

  v8f c[2][2] = {};
  for (int kb = 0; kb < K0; kb += 32) {
    v16bf a0 = load_a_frag(A0, K0, row0,      kb, lane);
    v16bf a1 = load_a_frag(A0, K0, row0 + 16, kb, lane);
    v16bf b0 = load_b_frag(Bt0, ldb0, col0,      kb, lane);
    v16bf b1 = load_b_frag(Bt0, ldb0, col0 + 16, kb, lane);
    c[0][0] = wmma_bf16(a0, b0, c[0][0]);
    c[0][1] = wmma_bf16(a0, b1, c[0][1]);
    c[1][0] = wmma_bf16(a1, b0, c[1][0]);
    c[1][1] = wmma_bf16(a1, b1, c[1][1]);
  }
  if (A1) {  // concat-K second operand pair
    for (int kb = 0; kb < K1; kb += 32) {
      v16bf a0 = load_a_frag(A1, K1, row0,      kb, lane);
      v16bf a1 = load_a_frag(A1, K1, row0 + 16, kb, lane);
      v16bf b0 = load_b_frag(Bt1, ldb1, col0,      kb, lane);
      v16bf b1 = load_b_frag(Bt1, ldb1, col0 + 16, kb, lane);
      c[0][0] = wmma_bf16(a0, b0, c[0][0]);
      c[0][1] = wmma_bf16(a0, b1, c[0][1]);
      c[1][0] = wmma_bf16(a1, b0, c[1][0]);
      c[1][1] = wmma_bf16(a1, b1, c[1][1]);
    }
  }
  const int mo = (lane & 16) ? 8 : 0;
  const int nn = lane & 15;
  #pragma unroll
  for (int i = 0; i < 2; ++i)
    #pragma unroll
    for (int j = 0; j < 2; ++j)
      #pragma unroll
      for (int r = 0; r < 8; ++r) {
        int m = row0 + 16 * i + r + mo;
        int n = col0 + 16 * j + nn;
        float v = c[i][j][r];
        if (bias) v += bias[n];
        if (act)  v = v > 0.f ? v : 0.f;
        if (outB) outB[(size_t)m * ldOut + n] = (__bf16)v;
        else      outF[(size_t)m * ldOut + n] = v;
      }
}

// ---- Big GEMM: out = sigmoid(z @ z^T), z bf16 [8192 x 512] -----------------
// Block 256 thr = 8 waves (4 M x 2 N), wave tile 64x64 (16 accumulators),
// block tile 256x128. B^T of z@z^T is z itself (N-major layout is free).
__global__ void __launch_bounds__(256)
zzt_sigmoid_kernel(const __bf16* __restrict__ z, float* __restrict__ out)
{
  const int lane = threadIdx.x & 31;
  const int wave = threadIdx.x >> 5;
  const int row0 = blockIdx.x * 256 + (wave & 3) * 64;
  const int col0 = blockIdx.y * 128 + (wave >> 2) * 64;

  v8f c[4][4] = {};
  for (int kb = 0; kb < 512; kb += 32) {
    v16bf a[4], b[4];
    #pragma unroll
    for (int i = 0; i < 4; ++i) a[i] = load_a_frag(z, 512, row0 + 16 * i, kb, lane);
    #pragma unroll
    for (int j = 0; j < 4; ++j) b[j] = load_b_frag(z, 512, col0 + 16 * j, kb, lane);
    #pragma unroll
    for (int i = 0; i < 4; ++i)
      #pragma unroll
      for (int j = 0; j < 4; ++j)
        c[i][j] = wmma_bf16(a[i], b[j], c[i][j]);
  }
  const int mo = (lane & 16) ? 8 : 0;
  const int nn = lane & 15;
  #pragma unroll
  for (int i = 0; i < 4; ++i)
    #pragma unroll
    for (int j = 0; j < 4; ++j)
      #pragma unroll
      for (int r = 0; r < 8; ++r) {
        int m = row0 + 16 * i + r + mo;
        int n = col0 + 16 * j + nn;
        float v = c[i][j][r];
        out[(size_t)m * NN + n] = 1.0f / (1.0f + __expf(-v));
      }
}

// ---- Elementwise / graph kernels ------------------------------------------

__global__ void zero_kernel(float* p, size_t n) {
  size_t i = (size_t)blockIdx.x * blockDim.x + threadIdx.x;
  if (i < n) p[i] = 0.f;
}

__global__ void cvt_bf16_kernel(const float* __restrict__ x, __bf16* __restrict__ y, size_t n) {
  size_t i = (size_t)blockIdx.x * blockDim.x + threadIdx.x;
  if (i < n) y[i] = (__bf16)x[i];
}

// W f32 [K x Ncols] row-major  ->  Wt bf16 [Ncols x K] (N-major for B frags)
__global__ void transpose_cvt_kernel(const float* __restrict__ W, __bf16* __restrict__ Wt,
                                     int K, int Ncols) {
  size_t idx = (size_t)blockIdx.x * blockDim.x + threadIdx.x;
  if (idx < (size_t)K * Ncols) {
    int k = (int)(idx / Ncols), n = (int)(idx % Ncols);
    Wt[(size_t)n * K + k] = (__bf16)W[idx];
  }
}

__global__ void deg_kernel(const int* __restrict__ dst, float* deg, int E) {
  int e = blockIdx.x * blockDim.x + threadIdx.x;
  if (e < E) unsafeAtomicAdd(&deg[dst[e]], 1.0f);
}

__global__ void dinv_kernel(float* d, int n) {  // in-place: deg -> rsqrt(deg+1)
  int i = blockIdx.x * blockDim.x + threadIdx.x;
  if (i < n) d[i] = rsqrtf(d[i] + 1.0f);       // +1 = self loop
}

// One wave per edge: agg[dst] += dinv[s]*dinv[d] * h[src], C channels.
__global__ void __launch_bounds__(256)
scatter_kernel(const float* __restrict__ h, const float* __restrict__ dinv,
               const int* __restrict__ src, const int* __restrict__ dst,
               float* __restrict__ agg, int C)
{
  int wave = threadIdx.x >> 5, lane = threadIdx.x & 31;
  int e = blockIdx.x * 8 + wave;
  int s = src[e], d = dst[e];
  float w = dinv[s] * dinv[d];
  const float* hs = h + (size_t)s * C;
  float* ad = agg + (size_t)d * C;
  for (int c = lane; c < C; c += 32)
    unsafeAtomicAdd(&ad[c], w * hs[c]);
}

// g = bf16( agg + dinv[i]^2 * h  (self loop)  + bias )
__global__ void assemble_kernel(const float* __restrict__ agg, const float* __restrict__ h,
                                const float* __restrict__ dinv, const float* __restrict__ bias,
                                __bf16* __restrict__ out, int C, size_t total)
{
  size_t idx = (size_t)blockIdx.x * blockDim.x + threadIdx.x;
  if (idx < total) {
    int i = (int)(idx / C), c = (int)(idx % C);
    float di = dinv[i];
    out[idx] = (__bf16)(agg[idx] + di * di * h[idx] + bias[c]);
  }
}

// ---------------------------------------------------------------------------

extern "C" void kernel_launch(void* const* d_in, const int* in_sizes, int n_in,
                              void* d_out, int out_size, void* d_ws, size_t ws_size,
                              hipStream_t stream) {
  const float* x_self     = (const float*)d_in[0];
  const float* x_neighbor = (const float*)d_in[1];
  const int*   edges      = (const int*)d_in[2];   // [2, E]: row0 src, row1 dst
  const float* w_in_self  = (const float*)d_in[3];
  const float* b_in_self  = (const float*)d_in[4];
  const float* w_out_self = (const float*)d_in[5];
  const float* b_out_self = (const float*)d_in[6];
  const float* gw1 = (const float*)d_in[7];
  const float* gb1 = (const float*)d_in[8];
  const float* gw2 = (const float*)d_in[9];
  const float* gb2 = (const float*)d_in[10];
  const float* w_out = (const float*)d_in[11];
  const float* b_out = (const float*)d_in[12];
  float* out = (float*)d_out;
  const int* src = edges;
  const int* dst = edges + EE;

  uint8_t* ws = (uint8_t*)d_ws;
  // Workspace layout (time-aliased; ~70 MB total)
  float*  dinv     = (float*)(ws + 0);                         // 32 KB (deg then dinv)
  __bf16* xself_bf = (__bf16*)(ws + 32768);                    // 8 MB
  __bf16* xnb_bf   = (__bf16*)(ws + 32768 + 8388608);          // 8 MB
  __bf16* wt       = (__bf16*)(ws + 32768 + 2 * 8388608);      // 2.13 MB of bf16 weights^T
  __bf16* wis_t = wt;            // [512 x 512]
  __bf16* wos_t = wt + 262144;   // [256 x 1024]
  __bf16* gw1_t = wt + 524288;   // [512 x 512]
  __bf16* gw2_t = wt + 786432;   // [256 x 512]
  __bf16* wo_t  = wt + 917504;   // [256 x 768]
  size_t oR1 = 32768 + 2 * 8388608 + 2228224;
  float*  R1     = (float*)(ws + oR1);                  // 16 MB: h1, then h2
  float*  R2     = (float*)(ws + oR1 + 16777216);       // 16 MB: l1a_bf, then agg1, then agg2
  __bf16* l1a_bf = (__bf16*)R2;                          // aliased; dead before agg1 zeroed
  __bf16* g1_bf  = (__bf16*)(ws + oR1 + 2 * 16777216);  // 8 MB
  __bf16* g2_bf  = (__bf16*)(ws + oR1 + 2 * 16777216 + 8388608);            // 4 MB
  __bf16* z_bf   = (__bf16*)(ws + oR1 + 2 * 16777216 + 8388608 + 4194304); // 8 MB [8192 x 512]

  const size_t n512 = (size_t)NN * 512, n256 = (size_t)NN * 256;

  // 1) degrees -> dinv
  zero_kernel<<<(NN + 255) / 256, 256, 0, stream>>>(dinv, NN);
  deg_kernel<<<(EE + 255) / 256, 256, 0, stream>>>(dst, dinv, EE);
  dinv_kernel<<<(NN + 255) / 256, 256, 0, stream>>>(dinv, NN);

  // 2) bf16 conversions + weight transposes
  cvt_bf16_kernel<<<(unsigned)(n512 / 256), 256, 0, stream>>>(x_self, xself_bf, n512);
  cvt_bf16_kernel<<<(unsigned)(n512 / 256), 256, 0, stream>>>(x_neighbor, xnb_bf, n512);
  transpose_cvt_kernel<<<262144 / 256, 256, 0, stream>>>(w_in_self,  wis_t, 512, 512);
  transpose_cvt_kernel<<<262144 / 256, 256, 0, stream>>>(w_out_self, wos_t, 1024, 256);
  transpose_cvt_kernel<<<262144 / 256, 256, 0, stream>>>(gw1, gw1_t, 512, 512);
  transpose_cvt_kernel<<<131072 / 256, 256, 0, stream>>>(gw2, gw2_t, 512, 256);
  transpose_cvt_kernel<<<196608 / 256, 256, 0, stream>>>(w_out, wo_t, 768, 256);

  // 3) linear branch: l1a = relu(x_self @ W_in + b)          [8192 x 512] bf16
  gemm_bf16_kernel<<<dim3(64, 8), 256, 0, stream>>>(
      xself_bf, wis_t, 512, 512, nullptr, nullptr, 0, 0,
      b_in_self, 1, nullptr, l1a_bf, 512);
  //    z[:, :256] = x_self@Wos[:512] + l1a@Wos[512:] + b      [8192 x 256] bf16
  gemm_bf16_kernel<<<dim3(64, 4), 256, 0, stream>>>(
      xself_bf, wos_t, 512, 1024, l1a_bf, wos_t + 512, 512, 1024,
      b_out_self, 0, nullptr, z_bf, 512);

  // 4) GCN layer 1: h1 = x_self @ gw1 (f32)
  gemm_bf16_kernel<<<dim3(64, 8), 256, 0, stream>>>(
      xself_bf, gw1_t, 512, 512, nullptr, nullptr, 0, 0,
      nullptr, 0, R1, nullptr, 512);
  zero_kernel<<<(unsigned)(n512 / 256), 256, 0, stream>>>(R2, n512);
  scatter_kernel<<<EE / 8, 256, 0, stream>>>(R1, dinv, src, dst, R2, 512);
  assemble_kernel<<<(unsigned)(n512 / 256), 256, 0, stream>>>(R2, R1, dinv, gb1, g1_bf, 512, n512);

  // 5) GCN layer 2: h2 = g1 @ gw2 (f32)
  gemm_bf16_kernel<<<dim3(64, 4), 256, 0, stream>>>(
      g1_bf, gw2_t, 512, 512, nullptr, nullptr, 0, 0,
      nullptr, 0, R1, nullptr, 256);
  zero_kernel<<<(unsigned)(n256 / 256), 256, 0, stream>>>(R2, n256);
  scatter_kernel<<<EE / 8, 256, 0, stream>>>(R1, dinv, src, dst, R2, 256);
  assemble_kernel<<<(unsigned)(n256 / 256), 256, 0, stream>>>(R2, R1, dinv, gb2, g2_bf, 256, n256);

  // 6) z[:, 256:] = x_nb@Wo[:512] + g2@Wo[512:] + b           [8192 x 256] bf16
  gemm_bf16_kernel<<<dim3(64, 4), 256, 0, stream>>>(
      xnb_bf, wo_t, 512, 768, g2_bf, wo_t + 512, 256, 768,
      b_out, 0, nullptr, z_bf + 256, 512);

  // 7) out = sigmoid(z @ z^T)   [8192 x 8192] f32
  zzt_sigmoid_kernel<<<dim3(NN / 256, NN / 128), 256, 0, stream>>>(z_bf, out);
}
